// DrugEncoderVAE_74053826117702
// MI455X (gfx1250) — matrix operand order repaired and checked by reading, not compiled, not run on hardware
//
#include <hip/hip_runtime.h>
#include <hip/hip_bf16.h>
#include <math.h>

// Problem constants (from reference)
#define BB    2048          // molecules
#define NPM   64            // atoms per molecule
#define DD    384           // feature dim
#define GD    128           // D/3
#define NN    (BB*NPM)      // 131072 nodes
#define EPM   128           // edges per molecule
#define AF    88            // atom features
#define AFP   96            // atom features padded to K%32==0
#define POUT  66            // conv output length: 64 + 2*pad - (k-1) = 66 for all three convs

typedef __bf16 bf16;
typedef __attribute__((ext_vector_type(16))) __bf16 v16bf;
typedef __attribute__((ext_vector_type(8)))  float  v8f;

template<int ACT> __device__ __forceinline__ float act_f(float v) {
  if constexpr (ACT == 1) return 0.5f * v * (1.f + erff(v * 0.70710678118654752f)); // exact gelu
  if constexpr (ACT == 2) return fmaxf(v, 0.f);
  if constexpr (ACT == 3) return tanhf(v);
  if constexpr (ACT == 4) return 1.f / (1.f + expf(-v));
  return v;
}

__device__ __forceinline__ v16bf cvt_frag(float4 f0, float4 f1, float4 f2, float4 f3) {
  v16bf a;
  a[0]=(bf16)f0.x;  a[1]=(bf16)f0.y;  a[2]=(bf16)f0.z;  a[3]=(bf16)f0.w;
  a[4]=(bf16)f1.x;  a[5]=(bf16)f1.y;  a[6]=(bf16)f1.z;  a[7]=(bf16)f1.w;
  a[8]=(bf16)f2.x;  a[9]=(bf16)f2.y;  a[10]=(bf16)f2.z; a[11]=(bf16)f2.w;
  a[12]=(bf16)f3.x; a[13]=(bf16)f3.y; a[14]=(bf16)f3.z; a[15]=(bf16)f3.w;
  return a;
}

// ---------------------------------------------------------------------------
// WMMA GEMM: out[M,Nc] = act(A[M,K] @ W[K,Nc] + bias).  K compile-time, K%32==0,
// M%128==0, Nc%32==0.  Block = 8 waves; block tile 128x32; wave tile 16x32.
// Whole Kx32 W panel is staged in LDS ONCE (bf16, transposed), then the K-loop
// is a pure ds_load -> cvt -> wmma stream with no barriers.
// A-frag (16-bit A 16x32): lane L, elem e -> M=L%16, K=e+16*(L/16)
//   => one contiguous 16-float run per lane: [kk+16h, +16)
// B-frag (16-bit B 32x16): lane L, elem e -> K=e+16*(L/16), N=L%16
//   => 16 contiguous bf16 in sB[n*K + kk + 16h], 32B-aligned (2K%32==0)
// C/D  (f32 16x16): lane L, vgpr r -> M=r+8*(L/16), N=L%16
// ---------------------------------------------------------------------------
template<int ACT, int K>
__global__ __launch_bounds__(256) void wmma_gemm(
    const float* __restrict__ A, const float* __restrict__ W,
    const float* __restrict__ bias, float* __restrict__ out, int Nc)
{
  __shared__ __align__(32) bf16 sB[K * 32];
  const int tid  = threadIdx.x;
  const int wave = tid >> 5, lane = tid & 31;
  const int lhalf = lane >> 4, lmod = lane & 15;
  const int row0 = (blockIdx.x * 8 + wave) * 16;
  const int col0 = blockIdx.y * 32;
  const float* __restrict__ Arow = A + (size_t)(row0 + lmod) * K;

  // Stage the whole W panel (K x 32) once: coalesced global reads, transposed store
  for (int idx = tid; idx < K * 32; idx += 256) {
    const int k = idx >> 5, nc = idx & 31;
    sB[nc * K + k] = (bf16)W[(size_t)k * Nc + col0 + nc];
  }
  __syncthreads();

  v8f acc0 = {0.f,0.f,0.f,0.f,0.f,0.f,0.f,0.f};
  v8f acc1 = {0.f,0.f,0.f,0.f,0.f,0.f,0.f,0.f};

  for (int kk = 0; kk < K; kk += 32) {
    __builtin_prefetch(&Arow[kk + 32], 0, 0);     // speculative, OOB-safe
    const float4 f0 = *(const float4*)&Arow[kk + 16 * lhalf];
    const float4 f1 = *(const float4*)&Arow[kk + 16 * lhalf + 4];
    const float4 f2 = *(const float4*)&Arow[kk + 16 * lhalf + 8];
    const float4 f3 = *(const float4*)&Arow[kk + 16 * lhalf + 12];
    const v16bf a  = cvt_frag(f0, f1, f2, f3);
    const v16bf b0 = *(const v16bf*)&sB[lmod * K + kk + 16 * lhalf];
    const v16bf b1 = *(const v16bf*)&sB[(lmod + 16) * K + kk + 16 * lhalf];
    acc0 = __builtin_amdgcn_wmma_f32_16x16x32_bf16(false, a, false, b0, (short)0, acc0, false, false);
    acc1 = __builtin_amdgcn_wmma_f32_16x16x32_bf16(false, a, false, b1, (short)0, acc1, false, false);
  }

  const float bv0 = bias[col0 + lmod];
  const float bv1 = bias[col0 + 16 + lmod];
#pragma unroll
  for (int r = 0; r < 8; ++r) {
    const int m = row0 + r + (lhalf << 3);
    out[(size_t)m * Nc + col0 + lmod]      = act_f<ACT>(acc0[r] + bv0);
    out[(size_t)m * Nc + col0 + 16 + lmod] = act_f<ACT>(acc1[r] + bv1);
  }
}

// ---------------------------------------------------------------------------
// Conv1d (NCH, zero-padded) as WMMA GEMM. M = BB*POUT rows, Nc = 128 channels,
// K = 128*KT (tap-major). Whole weight panel staged in LDS once. Sequence
// padding handled branch-free (clamped address + mask). ReLU fused.
// ---------------------------------------------------------------------------
template<int KT>
__global__ __launch_bounds__(256) void wmma_conv(
    const float* __restrict__ h, const float* __restrict__ cw,
    const float* __restrict__ cb, float* __restrict__ out, int gofs)
{
  constexpr int K = 128 * KT;
  __shared__ __align__(32) bf16 sB[K * 32];
  const int tid  = threadIdx.x;
  const int wave = tid >> 5, lane = tid & 31;
  const int lhalf = lane >> 4, lmod = lane & 15;
  const int row0 = (blockIdx.x * 8 + wave) * 16;
  const int col0 = blockIdx.y * 32;
  const int arow = row0 + lmod;
  const int ab = arow / POUT, ap = arow % POUT;
  constexpr int PAD = (KT == 1) ? 1 : (KT == 3 ? 2 : 3);   // reference pads 1/2/3

  // Stage weights: sB[nc*K + k] = cw[(col0+nc)*128 + i][t],  k = t*128 + i
  for (int idx = tid; idx < K * 32; idx += 256) {
    const int k = idx >> 5, nc = idx & 31;
    const int t = k >> 7, i = k & 127;
    sB[nc * K + k] = (bf16)cw[((size_t)(col0 + nc) * 128 + i) * KT + t];
  }
  __syncthreads();

  v8f acc0 = {0.f,0.f,0.f,0.f,0.f,0.f,0.f,0.f};
  v8f acc1 = {0.f,0.f,0.f,0.f,0.f,0.f,0.f,0.f};

  for (int kk = 0; kk < K; kk += 32) {
    const int t  = kk >> 7;          // tap index (uniform across the 32-wide K slab)
    const int i0 = kk & 127;         // input-channel offset within the tap
    const int pos = ap - PAD + t;
    const bool ok = (pos >= 0) && (pos < NPM);
    const float* hr = h + (size_t)(ab * NPM + (ok ? pos : 0)) * DD + gofs + i0;
    float4 f0 = *(const float4*)&hr[16 * lhalf];
    float4 f1 = *(const float4*)&hr[16 * lhalf + 4];
    float4 f2 = *(const float4*)&hr[16 * lhalf + 8];
    float4 f3 = *(const float4*)&hr[16 * lhalf + 12];
    const float msk = ok ? 1.f : 0.f;
    f0.x*=msk; f0.y*=msk; f0.z*=msk; f0.w*=msk;
    f1.x*=msk; f1.y*=msk; f1.z*=msk; f1.w*=msk;
    f2.x*=msk; f2.y*=msk; f2.z*=msk; f2.w*=msk;
    f3.x*=msk; f3.y*=msk; f3.z*=msk; f3.w*=msk;

    const v16bf a  = cvt_frag(f0, f1, f2, f3);
    const v16bf b0 = *(const v16bf*)&sB[lmod * K + kk + 16 * lhalf];
    const v16bf b1 = *(const v16bf*)&sB[(lmod + 16) * K + kk + 16 * lhalf];
    acc0 = __builtin_amdgcn_wmma_f32_16x16x32_bf16(false, a, false, b0, (short)0, acc0, false, false);
    acc1 = __builtin_amdgcn_wmma_f32_16x16x32_bf16(false, a, false, b1, (short)0, acc1, false, false);
  }

  const int o0 = col0 + lmod, o1 = col0 + 16 + lmod;
  const float bv0 = cb[o0], bv1 = cb[o1];
#pragma unroll
  for (int r = 0; r < 8; ++r) {
    const int m = row0 + r + (lhalf << 3);
    const int mb = m / POUT, mp = m % POUT;
    out[((size_t)mb * DD + gofs + o0) * POUT + mp] = fmaxf(acc0[r] + bv0, 0.f);
    out[((size_t)mb * DD + gofs + o1) * POUT + mp] = fmaxf(acc1[r] + bv1, 0.f);
  }
}

// ---------------------------------------------------------------------------
// Deterministic GIN aggregation (no float atomics): one wave per node scans its
// molecule's 128 edges; t[node] = h[node] + sum_{e: dst==node} (h[src] + ea@We + be)
// Channels handled as 3 float4 chunks per lane (fully coalesced).
// ---------------------------------------------------------------------------
__device__ __forceinline__ float4 ld4(const float* p) { return *(const float4*)p; }

__global__ __launch_bounds__(256) void gin_aggregate(
    const float* __restrict__ h, const int* __restrict__ eidx,
    const float* __restrict__ eattr, const float* __restrict__ ew,
    const float* __restrict__ eb, float* __restrict__ t)
{
  const int wave = threadIdx.x >> 5;
  const int lane = threadIdx.x & 31;
  const int node = blockIdx.x * 8 + wave;
  const int mol = node >> 6;
  const int c0 = lane * 4;                      // chunks at c0, c0+128, c0+256

  float4 acc[3];
#pragma unroll
  for (int j = 0; j < 3; ++j) acc[j] = ld4(&h[(size_t)node * DD + c0 + j * 128]);

  const int e0 = mol * EPM;
  for (int e = e0; e < e0 + EPM; ++e) {
    if (eidx[2 * e + 1] != node) continue;
    const int src = eidx[2 * e];
    const float a0 = eattr[6*e+0], a1 = eattr[6*e+1], a2 = eattr[6*e+2],
                a3 = eattr[6*e+3], a4 = eattr[6*e+4], a5 = eattr[6*e+5];
#pragma unroll
    for (int j = 0; j < 3; ++j) {
      const int ch = c0 + j * 128;
      const float4 hv = ld4(&h[(size_t)src * DD + ch]);
      const float4 bv = ld4(&eb[ch]);
      const float4 w0 = ld4(&ew[0*DD+ch]), w1 = ld4(&ew[1*DD+ch]), w2 = ld4(&ew[2*DD+ch]);
      const float4 w3 = ld4(&ew[3*DD+ch]), w4 = ld4(&ew[4*DD+ch]), w5 = ld4(&ew[5*DD+ch]);
      acc[j].x += hv.x + bv.x + a0*w0.x + a1*w1.x + a2*w2.x + a3*w3.x + a4*w4.x + a5*w5.x;
      acc[j].y += hv.y + bv.y + a0*w0.y + a1*w1.y + a2*w2.y + a3*w3.y + a4*w4.y + a5*w5.y;
      acc[j].z += hv.z + bv.z + a0*w0.z + a1*w1.z + a2*w2.z + a3*w3.z + a4*w4.z + a5*w5.z;
      acc[j].w += hv.w + bv.w + a0*w0.w + a1*w1.w + a2*w2.w + a3*w3.w + a4*w4.w + a5*w5.w;
    }
  }
#pragma unroll
  for (int j = 0; j < 3; ++j) *(float4*)&t[(size_t)node * DD + c0 + j * 128] = acc[j];
}

// ---- small VALU kernels ----------------------------------------------------
__global__ void pad_x(const float* __restrict__ x, float* __restrict__ xp) {
  int idx = blockIdx.x * blockDim.x + threadIdx.x;
  if (idx >= NN * AFP) return;
  int n = idx / AFP, k = idx % AFP;
  xp[idx] = (k < AF) ? x[(size_t)n * AF + k] : 0.f;
}

__global__ void mol_sum(const float* __restrict__ h, float* __restrict__ g) {
  const int b = blockIdx.x, ch = threadIdx.x;   // blockDim = 384
  float s = 0.f;
  for (int p = 0; p < NPM; ++p) s += h[((size_t)(b * NPM + p)) * DD + ch];
  g[(size_t)b * DD + ch] = s;
}

__global__ void transpose_seq(const float* __restrict__ cs, float* __restrict__ cT) {
  size_t idx = (size_t)blockIdx.x * blockDim.x + threadIdx.x;
  if (idx >= (size_t)BB * DD * POUT) return;
  int p = (int)(idx % POUT);
  size_t r = idx / POUT;
  int ch = (int)(r % DD);
  int b  = (int)(r / DD);
  cT[((size_t)b * POUT + p) * DD + ch] = cs[idx];
}

__global__ void attn_logits(const float* __restrict__ t1, const float* __restrict__ w2,
                            const float* __restrict__ b2, float* __restrict__ logits,
                            int rows) {
  const int wave = threadIdx.x >> 5, lane = threadIdx.x & 31;
  const int r = blockIdx.x * 8 + wave;
  if (r >= rows) return;
  float s = 0.f;
#pragma unroll
  for (int j = 0; j < 6; ++j) { int k = lane + j * 32; s += t1[(size_t)r * 192 + k] * w2[k]; }
  for (int off = 16; off > 0; off >>= 1) s += __shfl_xor(s, off, 32);
  if (lane == 0) logits[r] = s + b2[0];
}

__global__ void softmax_pos(const float* __restrict__ logits, float* __restrict__ sm) {
  __shared__ float red[128];
  const int b = blockIdx.x, p = threadIdx.x;   // blockDim = 128
  float v = (p < POUT) ? logits[b * POUT + p] : -INFINITY;
  red[p] = v; __syncthreads();
  for (int s = 64; s > 0; s >>= 1) { if (p < s) red[p] = fmaxf(red[p], red[p + s]); __syncthreads(); }
  const float mx = red[0]; __syncthreads();
  float ex = (p < POUT) ? expf(v - mx) : 0.f;
  red[p] = ex; __syncthreads();
  for (int s = 64; s > 0; s >>= 1) { if (p < s) red[p] += red[p + s]; __syncthreads(); }
  if (p < POUT) sm[b * POUT + p] = ex / red[0];
}

__global__ void attn_pool(const float* __restrict__ cT, const float* __restrict__ sm,
                          float* __restrict__ loc) {
  const int b = blockIdx.x, ch = threadIdx.x;  // blockDim = 384
  float s = 0.f;
  for (int p = 0; p < POUT; ++p) s += sm[b * POUT + p] * cT[((size_t)b * POUT + p) * DD + ch];
  loc[(size_t)b * DD + ch] = s;
}

__global__ void concat_fuse(const float* __restrict__ g, const float* __restrict__ loc,
                            float* __restrict__ fuse) {
  int idx = blockIdx.x * blockDim.x + threadIdx.x;
  if (idx >= BB * 2 * DD) return;
  int b = idx / (2 * DD), j = idx % (2 * DD);
  fuse[idx] = (j < DD) ? g[(size_t)b * DD + j] : loc[(size_t)b * DD + j - DD];
}

__global__ void gate_combine(const float* __restrict__ gate, const float* __restrict__ loc,
                             const float* __restrict__ g, float* __restrict__ fused) {
  int idx = blockIdx.x * blockDim.x + threadIdx.x;
  if (idx >= BB * DD) return;
  float gt = gate[idx];
  fused[idx] = gt * loc[idx] + (1.f - gt) * g[idx];
}

__global__ void vae_head(const float* __restrict__ mu, const float* __restrict__ lv,
                         const float* __restrict__ eps, float* __restrict__ out) {
  int idx = blockIdx.x * blockDim.x + threadIdx.x;
  if (idx >= BB * DD) return;
  const float m = mu[idx], l = lv[idx];
  out[idx] = m + eps[idx] * expf(0.5f * l);           // z
  out[(size_t)BB * DD + idx]     = m;                 // mu
  out[(size_t)2 * BB * DD + idx] = l;                 // logvar
}

// ---------------------------------------------------------------------------
extern "C" void kernel_launch(void* const* d_in, const int* in_sizes, int n_in,
                              void* d_out, int out_size, void* d_ws, size_t ws_size,
                              hipStream_t stream) {
  (void)in_sizes; (void)n_in; (void)out_size; (void)ws_size;

  const float* x       = (const float*)d_in[0];
  const int*   eidx    = (const int*)  d_in[1];
  const float* eattr   = (const float*)d_in[2];
  const float* eps     = (const float*)d_in[3];
  const float* atom_w  = (const float*)d_in[4];
  const float* atom_b  = (const float*)d_in[5];
  const float* gin_ew  = (const float*)d_in[6];
  const float* gin_eb  = (const float*)d_in[7];
  const float* gin_w1  = (const float*)d_in[8];
  const float* gin_b1  = (const float*)d_in[9];
  const float* gin_w2  = (const float*)d_in[10];
  const float* gin_b2  = (const float*)d_in[11];
  const float* c1w     = (const float*)d_in[12];
  const float* c1b     = (const float*)d_in[13];
  const float* c3w     = (const float*)d_in[14];
  const float* c3b     = (const float*)d_in[15];
  const float* c5w     = (const float*)d_in[16];
  const float* c5b     = (const float*)d_in[17];
  const float* attn_w1 = (const float*)d_in[18];
  const float* attn_b1 = (const float*)d_in[19];
  const float* attn_w2 = (const float*)d_in[20];
  const float* attn_b2 = (const float*)d_in[21];
  const float* gate_w1 = (const float*)d_in[22];
  const float* gate_b1 = (const float*)d_in[23];
  const float* gate_w2 = (const float*)d_in[24];
  const float* gate_b2 = (const float*)d_in[25];
  const float* mu_w    = (const float*)d_in[26];
  const float* mu_b    = (const float*)d_in[27];
  const float* lv_w    = (const float*)d_in[28];
  const float* lv_b    = (const float*)d_in[29];

  // Workspace: two big ping-pong buffers + small tensors
  const size_t BIG = (size_t)BB * DD * POUT;   // 51,904,512 floats (>= NN*DD and >= NN*AFP)
  float* ws     = (float*)d_ws;
  float* P      = ws;                 // current h
  float* Q      = ws + BIG;           // scratch / next h
  float* g_feat = ws + 2 * BIG;
  float* loc    = g_feat + (size_t)BB * DD;
  float* logits = loc    + (size_t)BB * DD;
  float* smax   = logits + (size_t)BB * POUT;
  float* fuse   = smax   + (size_t)BB * POUT;
  float* gate1  = fuse   + (size_t)BB * 2 * DD;
  float* gateb  = gate1  + (size_t)BB * DD;
  float* fusedv = gateb  + (size_t)BB * DD;
  float* muv    = fusedv + (size_t)BB * DD;
  float* lvv    = muv    + (size_t)BB * DD;

  const dim3 blk(256);

  // 1) atom embedding: pad x to K=96, then h = xp @ atom_w + atom_b
  pad_x<<<dim3((NN * AFP + 255) / 256), blk, 0, stream>>>(x, Q);
  wmma_gemm<0, AFP><<<dim3(NN / 128, DD / 32), blk, 0, stream>>>(Q, atom_w, atom_b, P, DD);

  // 2) 4 GIN layers (ping-pong P<->Q; after 4 layers h is back in ws[0])
  for (int l = 0; l < 4; ++l) {
    gin_aggregate<<<dim3(NN / 8), blk, 0, stream>>>(
        P, eidx, eattr, gin_ew + (size_t)l * 6 * DD, gin_eb + (size_t)l * DD, Q);
    wmma_gemm<1, DD><<<dim3(NN / 128, DD / 32), blk, 0, stream>>>(
        Q, gin_w1 + (size_t)l * DD * DD, gin_b1 + (size_t)l * DD, P, DD);
    wmma_gemm<0, DD><<<dim3(NN / 128, DD / 32), blk, 0, stream>>>(
        P, gin_w2 + (size_t)l * DD * DD, gin_b2 + (size_t)l * DD, Q, DD);
    float* t = P; P = Q; Q = t;
  }

  // 3) global sum pooling (reads h before it is recycled)
  mol_sum<<<dim3(BB), dim3(DD), 0, stream>>>(P, g_feat);

  // 4) multi-scale convs as WMMA GEMMs -> conv_seq (NCH, length 66) in Q
  wmma_conv<1><<<dim3(BB * POUT / 128, 4), blk, 0, stream>>>(P, c1w, c1b, Q, 0);
  wmma_conv<3><<<dim3(BB * POUT / 128, 4), blk, 0, stream>>>(P, c3w, c3b, Q, GD);
  wmma_conv<5><<<dim3(BB * POUT / 128, 4), blk, 0, stream>>>(P, c5w, c5b, Q, 2 * GD);

  // 5) transpose conv_seq -> cT [BB*POUT, DD] into P (h is dead now)
  transpose_seq<<<dim3((unsigned)((BIG + 255) / 256)), blk, 0, stream>>>(Q, P);

  // 6) attention: t1 = tanh(cT @ attn_w1 + b1) into Q, then logits/softmax/pool
  wmma_gemm<3, DD><<<dim3(BB * POUT / 128, 192 / 32), blk, 0, stream>>>(
      P, attn_w1, attn_b1, Q, 192);
  attn_logits<<<dim3(BB * POUT / 8), blk, 0, stream>>>(Q, attn_w2, attn_b2, logits, BB * POUT);
  softmax_pos<<<dim3(BB), dim3(128), 0, stream>>>(logits, smax);
  attn_pool<<<dim3(BB), dim3(DD), 0, stream>>>(P, smax, loc);

  // 7) gated fusion
  concat_fuse<<<dim3((BB * 2 * DD + 255) / 256), blk, 0, stream>>>(g_feat, loc, fuse);
  wmma_gemm<2, 2 * DD><<<dim3(BB / 128, DD / 32), blk, 0, stream>>>(fuse, gate_w1, gate_b1, gate1, DD);
  wmma_gemm<4, DD><<<dim3(BB / 128, DD / 32), blk, 0, stream>>>(gate1, gate_w2, gate_b2, gateb, DD);
  gate_combine<<<dim3((BB * DD + 255) / 256), blk, 0, stream>>>(gateb, loc, g_feat, fusedv);

  // 8) VAE head
  wmma_gemm<0, DD><<<dim3(BB / 128, DD / 32), blk, 0, stream>>>(fusedv, mu_w, mu_b, muv, DD);
  wmma_gemm<0, DD><<<dim3(BB / 128, DD / 32), blk, 0, stream>>>(fusedv, lv_w, lv_b, lvv, DD);
  vae_head<<<dim3((BB * DD + 255) / 256), blk, 0, stream>>>(muv, lvv, eps, (float*)d_out);
}
